// NonLocalBlock_44220983279675
// MI455X (gfx1250) — compile-verified, matrix operand order checked by reading, not run on hardware
//
#include <hip/hip_runtime.h>
#include <hip/hip_bf16.h>

typedef _Float16 v8h  __attribute__((ext_vector_type(8)));
typedef _Float16 v16h __attribute__((ext_vector_type(16)));
typedef float    v8f  __attribute__((ext_vector_type(8)));

constexpr int B = 4;
constexpr int C = 512;
constexpr int N = 4096;          // 64*64
constexpr int G = 32;            // groups
constexpr int CPG = C / G;       // 16 channels per group
// softmax scale folded into log2 domain: (1/sqrt(C)) * log2(e)
#define SCALE_L2E (0.04419417382415922f * 1.4426950408889634f)

// ---------------------------------------------------------------------------
// Fragment loader: 16(row) x 32(K) block, memory row-major along K.
// Works for WMMA f16 A-fragments (rows = M) and B-fragments (rows = N cols),
// per the CDNA5 16-bit operand VGPR layout:
//   lane 0-15  : row = lane,    elems 0-7 = K 0..7,  elems 8-15 = K 16..23
//   lane 16-31 : row = lane-16, elems 0-7 = K 8..15, elems 8-15 = K 24..31
// Two contiguous 16-byte loads per lane.
// ---------------------------------------------------------------------------
__device__ inline v16h load_frag(const _Float16* base, int row_stride, int lane) {
    int r  = lane & 15;
    int k0 = (lane < 16) ? 0 : 8;
    const _Float16* p = base + (size_t)r * row_stride + k0;
    v8h lo = *(const v8h*)(p);
    v8h hi = *(const v8h*)(p + 16);
    return __builtin_shufflevector(lo, hi, 0,1,2,3,4,5,6,7,8,9,10,11,12,13,14,15);
}

__device__ inline v8f wmma16x16x32(v16h a, v16h b, v8f c) {
    return __builtin_amdgcn_wmma_f32_16x16x32_f16(false, a, false, b, (short)0, c,
                                                  false, false);
}

// ---------------------------------------------------------------------------
// Double-buffered GEMM mainloop (M-tile = 32 rows, per-wave N-slice = 128).
// Two stages per k-step; each stage prefetches the NEXT fragment group and
// issues WMMAs on fragments fetched one stage earlier, so s_wait_loadcnt is
// always > 0 and fetch overlaps matrix issue.
// ---------------------------------------------------------------------------
__device__ inline void gemm_mainloop(const _Float16* __restrict__ A,
                                     const _Float16* __restrict__ W,
                                     int o0, int lane,
                                     v8f (&acc0)[8], v8f (&acc1)[8]) {
    const _Float16* A1 = A + (size_t)16 * C;
    v16h a0c = load_frag(A, C, lane);
    v16h a1c = load_frag(A1, C, lane);
    v16h bfc[4], bfn[4];
    #pragma unroll
    for (int of = 0; of < 4; ++of)
        bfc[of] = load_frag(W + (size_t)(o0 + of * 16) * C, C, lane);
    v16h a0n = a0c, a1n = a1c;
    for (int kk = 0; kk < C; kk += 32) {
        // stage 0: prefetch group 1 of this kk, consume group 0 (bfc)
        #pragma unroll
        for (int of = 0; of < 4; ++of)
            bfn[of] = load_frag(W + (size_t)(o0 + (4 + of) * 16) * C + kk, C, lane);
        #pragma unroll
        for (int of = 0; of < 4; ++of) {
            acc0[of] = wmma16x16x32(a0c, bfc[of], acc0[of]);
            acc1[of] = wmma16x16x32(a1c, bfc[of], acc1[of]);
        }
        // stage 1: prefetch group 0 + A fragments of kk+32, consume group 1 (bfn)
        if (kk + 32 < C) {
            #pragma unroll
            for (int of = 0; of < 4; ++of)
                bfc[of] = load_frag(W + (size_t)(o0 + of * 16) * C + kk + 32, C, lane);
            a0n = load_frag(A + kk + 32, C, lane);
            a1n = load_frag(A1 + kk + 32, C, lane);
        }
        #pragma unroll
        for (int of = 0; of < 4; ++of) {
            acc0[4 + of] = wmma16x16x32(a0c, bfn[of], acc0[4 + of]);
            acc1[4 + of] = wmma16x16x32(a1c, bfn[of], acc1[4 + of]);
        }
        a0c = a0n; a1c = a1n;
    }
}

// ---------------------------------------------------------------------------
// GroupNorm statistics: one block per (b,g); 16 channels x 4096 contiguous.
// ---------------------------------------------------------------------------
__global__ __launch_bounds__(256) void gn_stats_kernel(const float* __restrict__ x,
                                                       float* __restrict__ stats) {
    int bg = blockIdx.x;                 // b*G + g
    const float* xp = x + (size_t)bg * CPG * N;
    float s = 0.f, ss = 0.f;
    for (int i = threadIdx.x; i < CPG * N; i += 256) {
        float v = xp[i];
        s += v; ss += v * v;
    }
    __shared__ float red0[256], red1[256];
    red0[threadIdx.x] = s; red1[threadIdx.x] = ss;
    __syncthreads();
    for (int off = 128; off > 0; off >>= 1) {
        if ((int)threadIdx.x < off) {
            red0[threadIdx.x] += red0[threadIdx.x + off];
            red1[threadIdx.x] += red1[threadIdx.x + off];
        }
        __syncthreads();
    }
    if (threadIdx.x == 0) {
        float inv = 1.0f / (float)(CPG * N);
        float mean = red0[0] * inv;
        float var  = red1[0] * inv - mean * mean;
        stats[bg * 2 + 0] = mean;
        stats[bg * 2 + 1] = rsqrtf(var + 1e-6f);
    }
}

// ---------------------------------------------------------------------------
// GroupNorm apply + transpose [b][c][n] f32 -> [b][n][c] f16 via LDS tile.
// ---------------------------------------------------------------------------
__global__ __launch_bounds__(256) void gn_apply_kernel(const float* __restrict__ x,
                                                       const float* __restrict__ stats,
                                                       const float* __restrict__ gamma,
                                                       const float* __restrict__ beta,
                                                       _Float16* __restrict__ h16) {
    int b  = blockIdx.z;
    int c0 = blockIdx.x * 32;
    int n0 = blockIdx.y * 32;
    int tx = threadIdx.x & 31;           // n (read) / c (write)
    int ty = threadIdx.x >> 5;           // 0..7
    __shared__ float tile[32][33];
    #pragma unroll
    for (int r = 0; r < 4; ++r) {
        int c  = c0 + ty + 8 * r;
        int bg = b * G + (c >> 4);
        float mean = stats[bg * 2 + 0], rstd = stats[bg * 2 + 1];
        float v = x[((size_t)b * C + c) * N + n0 + tx];
        tile[ty + 8 * r][tx] = (v - mean) * rstd * gamma[c] + beta[c];
    }
    __syncthreads();
    #pragma unroll
    for (int r = 0; r < 4; ++r) {
        int n = n0 + ty + 8 * r;
        h16[((size_t)b * N + n) * C + c0 + tx] = (_Float16)tile[tx][ty + 8 * r];
    }
}

// ---------------------------------------------------------------------------
// Convert the four [C][C] f32 weight matrices to f16 (concatenated).
// ---------------------------------------------------------------------------
__global__ __launch_bounds__(256) void wcvt_kernel(const float* __restrict__ wq,
                                                   const float* __restrict__ wk,
                                                   const float* __restrict__ wv,
                                                   const float* __restrict__ wo,
                                                   _Float16* __restrict__ w16) {
    int i = blockIdx.x * 256 + threadIdx.x;
    int which = i / (C * C), j = i % (C * C);
    float v;
    if      (which == 0) v = wq[j];
    else if (which == 1) v = wk[j];
    else if (which == 2) v = wv[j];
    else                 v = wo[j];
    w16[i] = (_Float16)v;
}

// ---------------------------------------------------------------------------
// conv1x1 as GEMM: D[n][o] = h16[n][:] . W[o][:] + bias[o]
// vmode=0: store f16 [b][n][o] (q/k).  vmode=1: store f16 [b][o][n] (v).
// ---------------------------------------------------------------------------
__global__ __launch_bounds__(128) void conv1x1_kernel(const _Float16* __restrict__ h16,
                                                      const _Float16* __restrict__ w16,
                                                      const float* __restrict__ bias,
                                                      _Float16* __restrict__ out,
                                                      int vmode) {
    int b    = blockIdx.y;
    int i0   = blockIdx.x * 32;
    int lane = threadIdx.x & 31;
    int o0   = (threadIdx.x >> 5) * 128;
    const _Float16* A = h16 + ((size_t)b * N + i0) * C;
    v8f acc0[8] = {}, acc1[8] = {};
    gemm_mainloop(A, w16, o0, lane, acc0, acc1);
    int col = lane & 15, half = lane >> 4;
    auto store_tile = [&](v8f (&acc)[8], int ibase) {
        #pragma unroll
        for (int of = 0; of < 8; ++of) {
            int o = o0 + of * 16 + col;
            float bv = bias[o];
            if (vmode == 0) {
                #pragma unroll
                for (int e = 0; e < 8; ++e) {
                    int n = ibase + e + 8 * half;
                    out[((size_t)b * N + n) * C + o] = (_Float16)(acc[of][e] + bv);
                }
            } else {
                v8h pk;
                #pragma unroll
                for (int e = 0; e < 8; ++e) pk[e] = (_Float16)(acc[of][e] + bv);
                *(v8h*)(out + ((size_t)b * C + o) * N + ibase + 8 * half) = pk;
            }
        }
    };
    store_tile(acc0, i0);
    store_tile(acc1, i0 + 16);
}

// ---------------------------------------------------------------------------
// Flash attention: WG = 4 waves handles one 16-row query tile.
// Channel dim (512) split across waves for both QK^T (partials reduced via
// LDS) and the O accumulator (each wave keeps a 16x128 f32 slice in VGPRs).
// All 8 K fragments of a j-step are fetched as one clause before the QK^T
// WMMAs; V fragments are fetched BEFORE the softmax so their latency hides
// behind the barrier + VALU work; next K rows get a global_prefetch hint.
// ---------------------------------------------------------------------------
__global__ __launch_bounds__(128) void flash_attn_kernel(const _Float16* __restrict__ q16,
                                                         const _Float16* __restrict__ k16,
                                                         const _Float16* __restrict__ v16buf,
                                                         _Float16* __restrict__ o16) {
    int b    = blockIdx.y;
    int i0   = blockIdx.x * 16;
    int lane = threadIdx.x & 31;
    int w    = threadIdx.x >> 5;       // wave 0..3
    int c0   = w * 128;
    __shared__ float    lds_s[4][2][256];     // per-wave partial S tiles (8e x 32 lanes)
    __shared__ _Float16 lds_p[4][16][32];     // per-wave P staging (D-layout -> A-layout)

    const _Float16* Q = q16 + ((size_t)b * N + i0) * C + c0;
    const _Float16* K = k16 + (size_t)b * N * C + c0;
    const _Float16* V = v16buf + ((size_t)b * C + c0) * N;

    v16h qf[4];
    #pragma unroll
    for (int t = 0; t < 4; ++t) qf[t] = load_frag(Q + t * 32, C, lane);

    v8f ob[8] = {};
    float rm[8], rl[8];
    #pragma unroll
    for (int e = 0; e < 8; ++e) { rm[e] = -3.0e38f; rl[e] = 0.f; }
    int col = lane & 15, half = lane >> 4;

    for (int j = 0; j < N; j += 32) {
        // partial S over this wave's 128-channel slice: one 16-load clause,
        // then 8 WMMAs (two independent accumulator chains s0/s1).
        v16h kf[8];
        #pragma unroll
        for (int t = 0; t < 4; ++t) {
            kf[t]     = load_frag(K + (size_t)j * C + t * 32, C, lane);
            kf[4 + t] = load_frag(K + (size_t)(j + 16) * C + t * 32, C, lane);
        }
        v8f s0 = {}, s1 = {};
        #pragma unroll
        for (int t = 0; t < 4; ++t) {
            s0 = wmma16x16x32(qf[t], kf[t], s0);
            s1 = wmma16x16x32(qf[t], kf[4 + t], s1);
        }
        __syncthreads();                       // prior iteration's lds_s reads done
        #pragma unroll
        for (int e = 0; e < 8; ++e) {
            lds_s[w][0][e * 32 + lane] = s0[e];
            lds_s[w][1][e * 32 + lane] = s1[e];
        }
        // V fragments for THIS j-step: issue now, consume after softmax.
        v16h vf[8];
        #pragma unroll
        for (int cf = 0; cf < 8; ++cf)
            vf[cf] = load_frag(V + (size_t)(cf * 16) * N + j, N, lane);
        if (j + 32 < N)
            __builtin_prefetch(K + (size_t)(j + 32 + lane) * C, 0, 0);
        __syncthreads();
        float t0[8], t1[8];
        #pragma unroll
        for (int e = 0; e < 8; ++e) {
            int idx = e * 32 + lane;
            t0[e] = (lds_s[0][0][idx] + lds_s[1][0][idx] +
                     lds_s[2][0][idx] + lds_s[3][0][idx]) * SCALE_L2E;
            t1[e] = (lds_s[0][1][idx] + lds_s[1][1][idx] +
                     lds_s[2][1][idx] + lds_s[3][1][idx]) * SCALE_L2E;
        }
        // online softmax update (exp2 domain)
        float al[8];
        #pragma unroll
        for (int e = 0; e < 8; ++e) {
            float mx = fmaxf(t0[e], t1[e]);
            #pragma unroll
            for (int m = 1; m < 16; m <<= 1) mx = fmaxf(mx, __shfl_xor(mx, m, 16));
            float mn = fmaxf(rm[e], mx);
            al[e] = exp2f(rm[e] - mn);
            float p0 = exp2f(t0[e] - mn), p1 = exp2f(t1[e] - mn);
            t0[e] = p0; t1[e] = p1;
            float rs = p0 + p1;
            #pragma unroll
            for (int m = 1; m < 16; m <<= 1) rs += __shfl_xor(rs, m, 16);
            rl[e] = rl[e] * al[e] + rs;
            rm[e] = mn;
        }
        #pragma unroll
        for (int cf = 0; cf < 8; ++cf)
            #pragma unroll
            for (int e = 0; e < 8; ++e) ob[cf][e] *= al[e];
        // D-layout P -> A-layout f16 via this wave's private LDS staging
        #pragma unroll
        for (int e = 0; e < 8; ++e) {
            lds_p[w][8 * half + e][col]      = (_Float16)t0[e];
            lds_p[w][8 * half + e][16 + col] = (_Float16)t1[e];
        }
        v16h pf = load_frag(&lds_p[w][0][0], 32, lane);   // same-wave DS dep
        #pragma unroll
        for (int cf = 0; cf < 8; ++cf)
            ob[cf] = wmma16x16x32(pf, vf[cf], ob[cf]);
    }
    #pragma unroll
    for (int e = 0; e < 8; ++e) rl[e] = 1.0f / rl[e];
    #pragma unroll
    for (int cf = 0; cf < 8; ++cf) {
        int c = c0 + cf * 16 + col;
        #pragma unroll
        for (int e = 0; e < 8; ++e) {
            int i = i0 + e + 8 * half;
            o16[((size_t)b * N + i) * C + c] = (_Float16)(ob[cf][e] * rl[e]);
        }
    }
}

// ---------------------------------------------------------------------------
// Output projection + residual multiply (M-tile = 32):
// out[b][o][i] = x[b][o][i] * (O16[i][:] . Wo[o][:] + bo[o])
// D-layout columns give 8 consecutive i per lane -> coalesced f32 stores.
// ---------------------------------------------------------------------------
__global__ __launch_bounds__(128) void outproj_kernel(const _Float16* __restrict__ o16,
                                                      const _Float16* __restrict__ wo16,
                                                      const float* __restrict__ bo,
                                                      const float* __restrict__ x,
                                                      float* __restrict__ out) {
    int b    = blockIdx.y;
    int i0   = blockIdx.x * 32;
    int lane = threadIdx.x & 31;
    int o0   = (threadIdx.x >> 5) * 128;
    const _Float16* A = o16 + ((size_t)b * N + i0) * C;
    v8f acc0[8] = {}, acc1[8] = {};
    gemm_mainloop(A, wo16, o0, lane, acc0, acc1);
    int col = lane & 15, half = lane >> 4;
    auto store_tile = [&](v8f (&acc)[8], int ibase) {
        #pragma unroll
        for (int of = 0; of < 8; ++of) {
            int o = o0 + of * 16 + col;
            float bv = bo[o];
            size_t base = ((size_t)b * C + o) * N + ibase + 8 * half;
            #pragma unroll
            for (int e = 0; e < 8; ++e)
                out[base + e] = x[base + e] * (acc[of][e] + bv);
        }
    };
    store_tile(acc0, i0);
    store_tile(acc1, i0 + 16);
}

// ---------------------------------------------------------------------------
extern "C" void kernel_launch(void* const* d_in, const int* in_sizes, int n_in,
                              void* d_out, int out_size, void* d_ws, size_t ws_size,
                              hipStream_t stream) {
    const float* x     = (const float*)d_in[0];
    const float* gamma = (const float*)d_in[1];
    const float* beta  = (const float*)d_in[2];
    const float* wq    = (const float*)d_in[3];
    const float* bq    = (const float*)d_in[4];
    const float* wk    = (const float*)d_in[5];
    const float* bk    = (const float*)d_in[6];
    const float* wv    = (const float*)d_in[7];
    const float* bv    = (const float*)d_in[8];
    const float* wo    = (const float*)d_in[9];
    const float* bo    = (const float*)d_in[10];
    float* out = (float*)d_out;

    char* ws = (char*)d_ws;
    size_t off = 0;
    auto alloc = [&](size_t bytes) -> void* {
        void* p = ws + off;
        off += (bytes + 255) & ~(size_t)255;
        return p;
    };
    const size_t act_halves = (size_t)B * N * C;
    float*    stats  = (float*)   alloc((size_t)B * G * 2 * sizeof(float));
    _Float16* w16    = (_Float16*)alloc((size_t)4 * C * C * sizeof(_Float16));
    _Float16* h16    = (_Float16*)alloc(act_halves * sizeof(_Float16));
    _Float16* q16    = (_Float16*)alloc(act_halves * sizeof(_Float16));
    _Float16* k16    = (_Float16*)alloc(act_halves * sizeof(_Float16));
    _Float16* v16buf = (_Float16*)alloc(act_halves * sizeof(_Float16));
    _Float16* o16    = (_Float16*)alloc(act_halves * sizeof(_Float16));

    gn_stats_kernel<<<B * G, 256, 0, stream>>>(x, stats);
    gn_apply_kernel<<<dim3(C / 32, N / 32, B), 256, 0, stream>>>(x, stats, gamma, beta, h16);
    wcvt_kernel<<<(4 * C * C) / 256, 256, 0, stream>>>(wq, wk, wv, wo, w16);

    conv1x1_kernel<<<dim3(N / 32, B), 128, 0, stream>>>(h16, w16 + 0 * (size_t)C * C, bq, q16, 0);
    conv1x1_kernel<<<dim3(N / 32, B), 128, 0, stream>>>(h16, w16 + 1 * (size_t)C * C, bk, k16, 0);
    conv1x1_kernel<<<dim3(N / 32, B), 128, 0, stream>>>(h16, w16 + 2 * (size_t)C * C, bv, v16buf, 1);

    flash_attn_kernel<<<dim3(N / 16, B), 128, 0, stream>>>(q16, k16, v16buf, o16);

    outproj_kernel<<<dim3(N / 32, B), 128, 0, stream>>>(o16, w16 + 3 * (size_t)C * C, bo, x, out);
}
